// Recurrence_22196390986358
// MI455X (gfx1250) — matrix operand order repaired and checked by reading, not compiled, not run on hardware
//
#include <hip/hip_runtime.h>

typedef __attribute__((ext_vector_type(16))) __bf16 v16bf;
typedef __attribute__((ext_vector_type(8)))  float  v8f;

#define T_STEPS 16
#define NBATCH  256
#define PP      64
#define OO      64
#define HH      1024
#define EE      512
#define OBSD    128
#define NV      64
#define FTOT    36997   // 64 + 32768 + 1 + 1 + 1 + 64 + 4096 + 1 + 1

__device__ __forceinline__ unsigned short f2bf(float x) {
  unsigned u = __float_as_uint(x);
  u += 0x7FFFu + ((u >> 16) & 1u);          // round-to-nearest-even
  return (unsigned short)(u >> 16);
}
__device__ __forceinline__ float sigmoidf_(float x) { return 1.0f / (1.0f + expf(-x)); }

// async copy global -> LDS (16B per lane), tracked by ASYNCcnt
__device__ __forceinline__ void async_b128(unsigned lds_off, const void* gaddr) {
  asm volatile("global_load_async_to_lds_b128 %0, %1, off"
               :: "v"(lds_off), "v"(gaddr) : "memory");
}
__device__ __forceinline__ void wait_async0() {
  asm volatile("s_wait_asynccnt 0" ::: "memory");
}
__device__ __forceinline__ unsigned lds_off32(const void* p) {
  return (unsigned)(unsigned long long)p;   // low 32 bits of shared addr = LDS offset
}

// ---------------------------------------------------------------------------
// bf16 WMMA GEMM:  C[M x N] = act(A)[M x K] * W[N x K]^T + bias
// 256 threads (8 wave32), tile 128(M) x 64(N), K step 32, double-buffered LDS.
// ABF16=true : A already bf16 -> GLOBAL_LOAD_ASYNC_TO_LDS_B128 staging.
// ABF16=false: A f32 (+optional relu) -> load/convert/ds_store staging.
// Optional outputs: f32 C (ldc) and/or bf16 Cbf (ldcb, optional relu).
// Requires M%128==0, N%64==0, K%32==0 (true at every call site).
// ---------------------------------------------------------------------------
#define TM 128
#define TN 64
#define TK 32
#define LPAD 8   // LDS row = 40 ushorts = 80B (keeps 16B alignment)

template <bool ABF16>
__global__ __launch_bounds__(256)
void gemm_wmma(const void* __restrict__ Aptr, int reluA, long lda,
               const unsigned short* __restrict__ W, long ldw,
               const float* __restrict__ bias,
               float* __restrict__ C, long ldc,
               unsigned short* __restrict__ Cbf, long ldcb, int reluC,
               int K)
{
  __shared__ unsigned short sA[2][TM][TK + LPAD];   // 2 x 10240B
  __shared__ unsigned short sB[2][TN][TK + LPAD];   // 2 x  5120B

  const int tid  = threadIdx.x;
  const int wave = tid >> 5;
  const int lane = tid & 31;
  const long bm = (long)blockIdx.y * TM;
  const long bn = (long)blockIdx.x * TN;

  const int ar = tid >> 1;            // A tile row      (0..127)
  const int ac = (tid & 1) * 16;      // A tile col base (0/16)
  const int br = tid >> 2;            // B tile row      (0..63)
  const int bc = (tid & 3) * 8;       // B tile col base (0/8/16/24)

  const unsigned short* Abf = (const unsigned short*)Aptr + (bm + ar) * lda + ac;
  const float*          Af32 = (const float*)Aptr + (bm + ar) * lda + ac;
  const unsigned short* Wb   = W + (bn + br) * ldw + bc;

  v8f acc[4];
  #pragma unroll
  for (int i = 0; i < 4; ++i)
    #pragma unroll
    for (int r = 0; r < 8; ++r) acc[i][r] = 0.0f;

  auto stage = [&](int buf, long k0) {
    if (ABF16) {
      async_b128(lds_off32(&sA[buf][ar][ac]),     Abf + k0);
      async_b128(lds_off32(&sA[buf][ar][ac + 8]), Abf + k0 + 8);
      async_b128(lds_off32(&sB[buf][br][bc]),     Wb  + k0);
    } else {
      const float* Ap = Af32 + k0;
      #pragma unroll
      for (int q = 0; q < 4; ++q) {
        float4 f = ((const float4*)Ap)[q];
        if (reluA) {
          f.x = f.x > 0.f ? f.x : 0.f;  f.y = f.y > 0.f ? f.y : 0.f;
          f.z = f.z > 0.f ? f.z : 0.f;  f.w = f.w > 0.f ? f.w : 0.f;
        }
        sA[buf][ar][ac + q*4 + 0] = f2bf(f.x);
        sA[buf][ar][ac + q*4 + 1] = f2bf(f.y);
        sA[buf][ar][ac + q*4 + 2] = f2bf(f.z);
        sA[buf][ar][ac + q*4 + 3] = f2bf(f.w);
      }
      const unsigned short* Wp = Wb + k0;
      *(uint4*)&sB[buf][br][bc] = *(const uint4*)Wp;
      if (k0 + TK < K) __builtin_prefetch(Wp + TK, 0, 1);   // global_prefetch_b8
    }
  };

  // prologue: fill buffer 0
  stage(0, 0);
  if (ABF16) wait_async0();
  __syncthreads();

  int buf = 0;
  for (long k0 = 0; k0 < K; k0 += TK) {
    if (k0 + TK < K) stage(buf ^ 1, k0 + TK);    // prefetch next tile into other buffer

    // gather all fragments first, then issue the 4 WMMAs back-to-back
    const int mrow  = wave * 16 + (lane & 15);
    const int khalf = (lane >> 4) * 8;           // A: lanes 16-31 hold K+8 halves
    union { v16bf v; uint4 u[2]; } af;
    af.u[0] = *(const uint4*)&sA[buf][mrow][khalf];
    af.u[1] = *(const uint4*)&sA[buf][mrow][16 + khalf];

    const int kb = (lane >> 4) * 8;              // B: lanes 16-31 hold K+8 halves
    union { v16bf v; uint4 u[2]; } bfrag[4];
    #pragma unroll
    for (int nb = 0; nb < 4; ++nb) {
      const int bcol = nb * 16 + (lane & 15);
      bfrag[nb].u[0] = *(const uint4*)&sB[buf][bcol][kb];
      bfrag[nb].u[1] = *(const uint4*)&sB[buf][bcol][kb + 16];
    }
    #pragma unroll
    for (int nb = 0; nb < 4; ++nb)
      acc[nb] = __builtin_amdgcn_wmma_f32_16x16x32_bf16(
          false, af.v, false, bfrag[nb].v, (short)0, acc[nb], false, false);

    if (ABF16) wait_async0();                    // next tile landed in LDS
    __syncthreads();
    buf ^= 1;
  }

  // epilogue: C/D layout — lane col = lane&15, rows 8*(lane>>4)+r
  const long gm0 = bm + wave * 16 + ((lane >> 4) << 3);
  #pragma unroll
  for (int nb = 0; nb < 4; ++nb) {
    const long gn = bn + nb * 16 + (lane & 15);
    const float bv = bias ? bias[gn] : 0.0f;
    #pragma unroll
    for (int r = 0; r < 8; ++r) {
      const float val = acc[nb][r] + bv;
      if (C)   C[(gm0 + r) * ldc + gn] = val;
      if (Cbf) {
        const float w = (reluC && val < 0.f) ? 0.f : val;
        Cbf[(gm0 + r) * ldcb + gn] = f2bf(w);
      }
    }
  }
}

// ---------------------------------------------------------------------------
// Elementwise / helper kernels
// ---------------------------------------------------------------------------
__global__ void f32_to_bf16_kernel(const float* __restrict__ in,
                                   unsigned short* __restrict__ out, long n)
{
  long i = (long)blockIdx.x * blockDim.x + threadIdx.x;
  if (i < n) out[i] = f2bf(in[i]);
}

__global__ __launch_bounds__(256)
void gather_relu_bf16_kernel(const int* __restrict__ obs,
                             const float* __restrict__ table,
                             unsigned short* __restrict__ out)
{
  const long row = blockIdx.x;                       // 0 .. T*N-1
  const int* orow = obs + row * OBSD;
  unsigned short* dst = out + row * (long)(OBSD * NV);
  for (int idx = threadIdx.x; idx < OBSD * NV; idx += 256) {
    const int d = idx >> 6, k = idx & 63;
    float x = table[orow[d] * NV + k];
    x = x > 0.f ? x : 0.f;
    dst[idx] = f2bf(x);
  }
}

__global__ void zero_kernel(float* p, int n)
{
  int i = blockIdx.x * blockDim.x + threadIdx.x;
  if (i < n) p[i] = 0.0f;
}

__global__ void copy_strided_kernel(const float* __restrict__ src, long sstride,
                                    float* __restrict__ dst, long dstride,
                                    int rows, int cols)
{
  int i = blockIdx.x * blockDim.x + threadIdx.x;
  if (i >= rows * cols) return;
  int r = i / cols, c = i % cols;
  dst[(long)r * dstride + c] = src[(long)r * sstride + c];
}

// x_in = [state, opt_table[opt_j]]  stored directly as bf16 for the GRU GEMM
__global__ void xin_kernel(const float* __restrict__ state, long sstride,
                           const float* __restrict__ opt_table,
                           const int* __restrict__ options, int j,
                           unsigned short* __restrict__ xin)
{
  int i = blockIdx.x * blockDim.x + threadIdx.x;
  if (i >= NBATCH * (EE + OO)) return;
  int n = i / (EE + OO), c = i % (EE + OO);
  float v = (c < EE) ? state[(long)n * sstride + c]
                     : opt_table[options[n * PP + j] * OO + (c - EE)];
  xin[i] = f2bf(v);
}

// GRU with zero hidden:  gh == b_hh  =>  h = (1 - z) * tanh(inn + r*b_hh_n)
// Emits bf16 (optionally relu'd) so the next GEMM can use the async-LDS path.
__global__ void gru_zero_h_kernel(const float* __restrict__ gi,
                                  const float* __restrict__ bhh,
                                  unsigned short* __restrict__ h, int reluOut)
{
  int i = blockIdx.x * blockDim.x + threadIdx.x;
  if (i >= NBATCH * HH) return;
  const int n = i >> 10, c = i & (HH - 1);
  const float* g = gi + (long)n * (3 * HH);
  const float r  = sigmoidf_(g[c]          + bhh[c]);
  const float z  = sigmoidf_(g[HH + c]     + bhh[HH + c]);
  const float nn = tanhf    (g[2 * HH + c] + r * bhh[2 * HH + c]);
  float v = (1.0f - z) * nn;
  if (reluOut && v < 0.f) v = 0.f;
  h[i] = f2bf(v);
}

// sharpness + log_softmax + lp/ent accumulation (one block of 64 per batch row)
__global__ __launch_bounds__(64)
void policy_kernel(const float* __restrict__ state, long sstride,
                   const float* __restrict__ W_sharp, const float* __restrict__ b_sharp,
                   const float* __restrict__ v, long vstride,
                   const int* __restrict__ options, int j,
                   float* __restrict__ lp, float* __restrict__ ent)
{
  __shared__ float red[64];
  __shared__ float lpv[64];
  const int n = blockIdx.x, tid = threadIdx.x;
  const float* s = state + (long)n * sstride;

  float acc = 0.f;
  for (int c = tid; c < EE; c += 64) {
    float x = s[c]; x = x > 0.f ? x : 0.f;
    acc += x * W_sharp[c];
  }
  red[tid] = acc;
  for (int off = 32; off > 0; off >>= 1) { __syncthreads(); if (tid < off) red[tid] += red[tid + off]; }
  __syncthreads();
  const float sp = red[0] + b_sharp[0];
  const float sharp = sp > 20.f ? sp : log1pf(expf(sp));
  __syncthreads();

  const float vo = v[(long)n * vstride + tid];
  const float l  = sharp * vo;
  red[tid] = l;
  for (int off = 32; off > 0; off >>= 1) { __syncthreads(); if (tid < off) red[tid] = fmaxf(red[tid], red[tid + off]); }
  __syncthreads();
  const float mx = red[0];
  __syncthreads();
  red[tid] = expf(l - mx);
  for (int off = 32; off > 0; off >>= 1) { __syncthreads(); if (tid < off) red[tid] += red[tid + off]; }
  __syncthreads();
  const float logZ = mx + logf(red[0]);
  __syncthreads();
  const float logp = l - logZ;
  lpv[tid] = logp;
  red[tid] = expf(logp) * logp;
  for (int off = 32; off > 0; off >>= 1) { __syncthreads(); if (tid < off) red[tid] += red[tid + off]; }
  __syncthreads();
  if (tid == 0) {
    lp[n]  += lpv[options[n * PP + j]];
    ent[n] -= red[0];
  }
}

__global__ __launch_bounds__(128)
void model_loss_kernel(const float* __restrict__ states_seq,
                       const float* __restrict__ emb_all,
                       float* __restrict__ mloss)
{
  __shared__ float red[128];
  const int row = blockIdx.x;                  // t*N + n
  const int t = row / NBATCH, n = row % NBATCH;
  const float* st = states_seq + ((long)n * PP + t) * EE;
  const float* tg = emb_all + (long)row * EE;
  float acc = 0.f;
  for (int e = threadIdx.x; e < EE; e += 128) { float d = st[e] - tg[e]; acc += d * d; }
  red[threadIdx.x] = acc;
  for (int off = 64; off > 0; off >>= 1) { __syncthreads(); if (threadIdx.x < off) red[threadIdx.x] += red[threadIdx.x + off]; }
  if (threadIdx.x == 0) mloss[row] = red[0] * (1.0f / EE);
}

// Broadcast pack: out[(t*N+n)*F + f]; t in 0..16, slice 16 mirrors t=15.
__global__ __launch_bounds__(256)
void pack_kernel(const float* __restrict__ states_seq,   // (N,P,E)
                 const float* __restrict__ values_seq,   // (N,P,O)
                 const float* __restrict__ mloss,        // (T,N)
                 const float* __restrict__ lp, const float* __restrict__ ent,
                 const int* __restrict__ options,        // (N,P)
                 float* __restrict__ out)
{
  const int row = blockIdx.x;                  // 0 .. 17*N-1
  const int t = row / NBATCH, n = row % NBATCH;
  const int teff = (t == T_STEPS) ? (T_STEPS - 1) : t;
  float* o = out + (long)row * FTOT;
  const float* st = states_seq + (long)n * (PP * EE);
  const float* va = values_seq + (long)n * (PP * OO);
  const int*   op = options + n * PP;
  for (int f = threadIdx.x; f < FTOT; f += 256) {
    float val;
    if (f < 64)                       val = (float)f;                         // indices
    else if (f < 64 + PP * EE)        val = st[f - 64];                       // states
    else if (f == 32832)              val = mloss[teff * NBATCH + n];         // model_loss
    else if (f == 32833)              val = lp[n];                            // log_probs
    else if (f == 32834)              val = ent[n];                           // entropy
    else if (f < 32835 + PP)          val = (float)op[f - 32835];             // options
    else if (f < 32899 + PP * OO)     val = va[f - 32899];                    // values
    else if (f == 36995)              val = (float)op[teff];                  // a
    else                              val = va[teff * OO + op[teff]];         // v_sel
    o[f] = val;
  }
}

// ---------------------------------------------------------------------------
extern "C" void kernel_launch(void* const* d_in, const int* in_sizes, int n_in,
                              void* d_out, int out_size, void* d_ws, size_t ws_size,
                              hipStream_t stream) {
  (void)in_sizes; (void)n_in; (void)out_size; (void)ws_size;
  const int*   obs      = (const int*)  d_in[0];
  const int*   options  = (const int*)  d_in[1];
  const float* emb1_tab = (const float*)d_in[2];
  const float* W_e0     = (const float*)d_in[3];
  const float* b_e0     = (const float*)d_in[4];
  const float* W_e1     = (const float*)d_in[5];
  const float* b_e1     = (const float*)d_in[6];
  const float* W_emb2   = (const float*)d_in[7];
  const float* b_emb2   = (const float*)d_in[8];
  const float* W_sharp  = (const float*)d_in[9];
  const float* b_sharp  = (const float*)d_in[10];
  const float* W_crit   = (const float*)d_in[11];
  const float* b_crit   = (const float*)d_in[12];
  const float* opt_tab  = (const float*)d_in[13];
  const float* W_ih0    = (const float*)d_in[14];
  // d_in[15] = W_hh0 (unused: hidden state is always zero)
  const float* b_ih0    = (const float*)d_in[16];
  const float* b_hh0    = (const float*)d_in[17];
  const float* W_ih1    = (const float*)d_in[18];
  // d_in[19] = W_hh1 (unused)
  const float* b_ih1    = (const float*)d_in[20];
  const float* b_hh1    = (const float*)d_in[21];
  float* out = (float*)d_out;

  // ---- workspace bump allocator ----
  char* ws = (char*)d_ws;
  size_t off = 0;
  auto alloc = [&](size_t bytes) -> void* {
    void* p = ws + off;
    off = (off + bytes + 255) & ~(size_t)255;
    return p;
  };
  const long TN_ = (long)T_STEPS * NBATCH;                 // 4096
  unsigned short* wb_e0   = (unsigned short*)alloc((size_t)HH * (NV*OBSD) * 2);
  unsigned short* wb_e1   = (unsigned short*)alloc((size_t)HH * HH * 2);
  unsigned short* wb_emb2 = (unsigned short*)alloc((size_t)EE * HH * 2);
  unsigned short* wb_ih0  = (unsigned short*)alloc((size_t)(3*HH) * (EE+OO) * 2);
  unsigned short* wb_ih1  = (unsigned short*)alloc((size_t)(3*HH) * HH * 2);
  unsigned short* wb_crit = (unsigned short*)alloc((size_t)OO * EE * 2);
  unsigned short* Ag      = (unsigned short*)alloc((size_t)TN_ * (NV*OBSD) * 2); // 67MB
  unsigned short* e0b     = (unsigned short*)alloc((size_t)TN_ * HH * 2);
  float* emb_all    = (float*)alloc((size_t)TN_ * EE * 4);
  float* states_seq = (float*)alloc((size_t)NBATCH * PP * EE * 4);
  float* values_seq = (float*)alloc((size_t)NBATCH * PP * OO * 4);
  unsigned short* xinb = (unsigned short*)alloc((size_t)NBATCH * (EE+OO) * 2);
  float* gi         = (float*)alloc((size_t)NBATCH * 3 * HH * 4);
  unsigned short* h0b = (unsigned short*)alloc((size_t)NBATCH * HH * 2);
  unsigned short* h1b = (unsigned short*)alloc((size_t)NBATCH * HH * 2);
  float* dummy      = (float*)alloc((size_t)NBATCH * EE * 4);
  float* lp         = (float*)alloc((size_t)NBATCH * 4);
  float* ent        = (float*)alloc((size_t)NBATCH * 4);
  float* mloss      = (float*)alloc((size_t)TN_ * 4);
  // Ag is dead after GEMM1 -> recycle its space for e1 (bf16)
  unsigned short* e1b = Ag;

  // ---- per-call init ----
  zero_kernel<<<1, 256, 0, stream>>>(lp, NBATCH);
  zero_kernel<<<1, 256, 0, stream>>>(ent, NBATCH);

  // ---- weight f32 -> bf16 ----
  auto cvt = [&](const float* src, unsigned short* dst, long n) {
    f32_to_bf16_kernel<<<(unsigned)((n + 255) / 256), 256, 0, stream>>>(src, dst, n);
  };
  cvt(W_e0,   wb_e0,   (long)HH * NV * OBSD);
  cvt(W_e1,   wb_e1,   (long)HH * HH);
  cvt(W_emb2, wb_emb2, (long)EE * HH);
  cvt(W_ih0,  wb_ih0,  (long)3 * HH * (EE + OO));
  cvt(W_ih1,  wb_ih1,  (long)3 * HH * HH);
  cvt(W_crit, wb_crit, (long)OO * EE);

  // ---- phase 1: embedding trunk (bf16 activations end-to-end) ----
  gather_relu_bf16_kernel<<<(unsigned)TN_, 256, 0, stream>>>(obs, emb1_tab, Ag);
  gemm_wmma<true><<<dim3(HH / TN, TN_ / TM), 256, 0, stream>>>(
      Ag, 0, NV * OBSD, wb_e0, NV * OBSD, b_e0,
      nullptr, 0, e0b, HH, 1, NV * OBSD);                          // e0 = relu -> bf16
  gemm_wmma<true><<<dim3(HH / TN, TN_ / TM), 256, 0, stream>>>(
      e0b, 0, HH, wb_e1, HH, b_e1,
      nullptr, 0, e1b, HH, 1, HH);                                 // e1 = relu -> bf16
  gemm_wmma<true><<<dim3(EE / TN, TN_ / TM), 256, 0, stream>>>(
      e1b, 0, HH, wb_emb2, HH, b_emb2,
      emb_all, EE, nullptr, 0, 0, HH);                             // emb_all f32
  // state_0 = emb_all[0]  -> states_seq[:,0,:]
  copy_strided_kernel<<<(NBATCH * EE + 255) / 256, 256, 0, stream>>>(
      emb_all, EE, states_seq, (long)PP * EE, NBATCH, EE);

  // ---- phase 2: sequential planning scan (P = 64 steps) ----
  for (int j = 0; j < PP; ++j) {
    const float* state = states_seq + (long)j * EE;              // row stride P*E
    // critic: v = relu(state) @ W_crit^T + b_crit  -> values_seq[:, j, :]
    gemm_wmma<false><<<dim3(OO / TN, NBATCH / TM), 256, 0, stream>>>(
        state, 1, (long)PP * EE, wb_crit, EE, b_crit,
        values_seq + (long)j * OO, (long)PP * OO, nullptr, 0, 0, EE);
    policy_kernel<<<NBATCH, 64, 0, stream>>>(
        state, (long)PP * EE, W_sharp, b_sharp,
        values_seq + (long)j * OO, (long)PP * OO, options, j, lp, ent);
    // x_in = [state, opt_table[opt_j]]  (bf16)
    xin_kernel<<<(NBATCH * (EE + OO) + 255) / 256, 256, 0, stream>>>(
        state, (long)PP * EE, opt_tab, options, j, xinb);
    // GRU cell 0 (zero hidden)
    gemm_wmma<true><<<dim3(3 * HH / TN, NBATCH / TM), 256, 0, stream>>>(
        xinb, 0, EE + OO, wb_ih0, EE + OO, b_ih0,
        gi, 3 * HH, nullptr, 0, 0, EE + OO);
    gru_zero_h_kernel<<<(NBATCH * HH + 255) / 256, 256, 0, stream>>>(gi, b_hh0, h0b, 0);
    // GRU cell 1 (zero hidden)
    gemm_wmma<true><<<dim3(3 * HH / TN, NBATCH / TM), 256, 0, stream>>>(
        h0b, 0, HH, wb_ih1, HH, b_ih1,
        gi, 3 * HH, nullptr, 0, 0, HH);
    gru_zero_h_kernel<<<(NBATCH * HH + 255) / 256, 256, 0, stream>>>(gi, b_hh1, h1b, 1); // relu'd
    // new_state = relu(h1) @ W_emb2^T + b_emb2 -> states_seq[:, j+1, :] (last discarded)
    float* cdst = (j < PP - 1) ? (states_seq + (long)(j + 1) * EE) : dummy;
    long   cldc = (j < PP - 1) ? (long)PP * EE : EE;
    gemm_wmma<true><<<dim3(EE / TN, NBATCH / TM), 256, 0, stream>>>(
        h1b, 0, HH, wb_emb2, HH, b_emb2,
        cdst, cldc, nullptr, 0, 0, HH);
  }

  // ---- phase 3: losses + packed output ----
  model_loss_kernel<<<(unsigned)TN_, 128, 0, stream>>>(states_seq, emb_all, mloss);
  pack_kernel<<<(T_STEPS + 1) * NBATCH, 256, 0, stream>>>(
      states_seq, values_seq, mloss, lp, ent, options, out);
}
